// Attention_66623532696044
// MI455X (gfx1250) — compile-verified, matrix operand order checked by reading, not run on hardware
//
#include <hip/hip_runtime.h>

// ---------------------------------------------------------------------------
// Attention (B=1, N=4096, C=768, H=12, D=64) for gfx1250 / MI455X.
// All GEMMs via v_wmma_f32_16x16x32_bf16 (wave32), flash-attention softmax,
// K/V tiles staged through LDS with global_load_async_to_lds_b128 (ASYNCcnt),
// double-buffered.
// ---------------------------------------------------------------------------

#define DEVINL __device__ __forceinline__

typedef __attribute__((ext_vector_type(8)))  __bf16 v8bf;
typedef __attribute__((ext_vector_type(16))) __bf16 v16bf;
typedef __attribute__((ext_vector_type(8)))  float  v8f;

union V8F { v8f v; float f[8]; };

static constexpr int Nseq = 4096;
static constexpr int C    = 768;
static constexpr int H    = 12;
static constexpr int Dh   = 64;
static constexpr int F3   = 3 * C;           // 2304
static constexpr float SCALE = 0.125f;       // 64^-0.5

DEVINL __bf16 f32_to_bf16(float f) {
  unsigned u = __builtin_bit_cast(unsigned, f);
  u += 0x7FFFu + ((u >> 16) & 1u);           // round-to-nearest-even
  unsigned short h = (unsigned short)(u >> 16);
  return __builtin_bit_cast(__bf16, h);
}

DEVINL v8f wmma_bf16(v16bf a, v16bf b, v8f c) {
  // D(16x16,f32) = A(16x32,bf16) * B(32x16,bf16) + C
  return __builtin_amdgcn_wmma_f32_16x16x32_bf16(
      /*neg_a=*/false, a, /*neg_b=*/false, b,
      /*c_mod=*/(short)0, c, /*reuse_a=*/false, /*reuse_b=*/false);
}

// Row-fragment loader (16-bit A layout, ISA 7.12.2):
//   lane L (0..15) : row = row0+L,    K = {k0+0..7,  k0+16..23}
//   lane L (16..31): row = row0+L-16, K = {k0+8..15, k0+24..31}
// The same layout mirrored with N in the lane slot is used for B operands,
// so "rows of B-transposed" load with this one helper. Works for global or
// LDS-resident source (ds_load_b128 after addrspace inference).
DEVINL v16bf load_frag(const __bf16* base, int ld, int row0, int k0, int lane) {
  int r  = row0 + (lane & 15);
  int kk = k0 + ((lane & 16) >> 1);          // +8 for upper half-wave
  const __bf16* p = base + (size_t)r * ld + kk;
  v8bf lo = *(const v8bf*)p;                 // 16B -> *_load_b128
  v8bf hi = *(const v8bf*)(p + 16);
  return __builtin_shufflevector(lo, hi, 0,1,2,3,4,5,6,7,8,9,10,11,12,13,14,15);
}

// Async copy of one 16B chunk per lane into LDS (tracked by ASYNCcnt).
DEVINL void async_ld_b128(unsigned lds_byte_off, const __bf16* gptr) {
  asm volatile("global_load_async_to_lds_b128 %0, %1, off"
               :: "v"(lds_byte_off), "v"(gptr)
               : "memory");
}
DEVINL void wait_async0() {
  asm volatile("s_wait_asynccnt 0x0" ::: "memory");
}

// reductions across a 16-lane half-wave (one matrix row of the C/D layout)
DEVINL float redmax16(float x) {
  x = fmaxf(x, __shfl_xor(x, 1, 32));
  x = fmaxf(x, __shfl_xor(x, 2, 32));
  x = fmaxf(x, __shfl_xor(x, 4, 32));
  x = fmaxf(x, __shfl_xor(x, 8, 32));
  return x;
}
DEVINL float redsum16(float x) {
  x += __shfl_xor(x, 1, 32);
  x += __shfl_xor(x, 2, 32);
  x += __shfl_xor(x, 4, 32);
  x += __shfl_xor(x, 8, 32);
  return x;
}

// ---------------------------------------------------------------------------
// fp32 -> bf16 conversion (grid-stride)
// ---------------------------------------------------------------------------
__global__ void cvt_bf16_kernel(const float* __restrict__ src,
                                __bf16* __restrict__ dst, int n) {
  int i = blockIdx.x * blockDim.x + threadIdx.x;
  int stride = gridDim.x * blockDim.x;
  for (; i < n; i += stride) dst[i] = f32_to_bf16(src[i]);
}

// ---------------------------------------------------------------------------
// QKV projection: qkv[n][f] = sum_c x[n][c] * Wqkv[f][c]
// Each wave computes a 16(M) x 64(F) strip. Output scattered into
// q[H][N][Dh], k[H][N][Dh], vT[H][Dh][N] as bf16.
// ---------------------------------------------------------------------------
__global__ void qkv_gemm_kernel(const __bf16* __restrict__ xb,
                                const __bf16* __restrict__ wb,
                                __bf16* __restrict__ qb,
                                __bf16* __restrict__ kb,
                                __bf16* __restrict__ vT) {
  const int lane = threadIdx.x & 31;
  const int wid  = blockIdx.x * (blockDim.x >> 5) + (threadIdx.x >> 5);
  const int mt = wid % (Nseq / 16);          // 256 M tiles
  const int ft = wid / (Nseq / 16);          // 36 F strips of 64
  const int m0 = mt * 16;
  const int f0 = ft * 64;

  V8F acc[4];
#pragma unroll
  for (int t = 0; t < 4; ++t)
#pragma unroll
    for (int j = 0; j < 8; ++j) acc[t].f[j] = 0.f;

  for (int kk = 0; kk < C; kk += 32) {
    v16bf a = load_frag(xb, C, m0, kk, lane);
#pragma unroll
    for (int t = 0; t < 4; ++t) {
      v16bf b = load_frag(wb, C, f0 + t * 16, kk, lane);
      acc[t].v = wmma_bf16(a, b, acc[t].v);
    }
  }

  const int sIdx = f0 / C;                   // 0=q 1=k 2=v (uniform per wave)
  const int h    = (f0 % C) / Dh;            // head (uniform; strips align)
#pragma unroll
  for (int t = 0; t < 4; ++t) {
#pragma unroll
    for (int j = 0; j < 8; ++j) {
      int n = m0 + j + ((lane >> 4) << 3);
      int d = t * 16 + (lane & 15);
      __bf16 val = f32_to_bf16(acc[t].f[j]);
      if (sIdx == 0)      qb[((size_t)h * Nseq + n) * Dh + d] = val;
      else if (sIdx == 1) kb[((size_t)h * Nseq + n) * Dh + d] = val;
      else                vT[((size_t)h * Dh + d) * Nseq + n] = val;
    }
  }
}

// ---------------------------------------------------------------------------
// Flash attention: block = (head, 64-query tile) = 4 waves x 16 queries.
// K tile (32x64, 4KB contiguous) and V^T tile (64x32, 4KB) are async-staged
// into LDS once per block, double-buffered. Per 32 keys per wave:
// 4 QK^T WMMAs -> online softmax (fp32) -> P via LDS -> 4 PV WMMAs.
// ---------------------------------------------------------------------------
__global__ void flash_attn_kernel(const __bf16* __restrict__ qb,
                                  const __bf16* __restrict__ kb,
                                  const __bf16* __restrict__ vT,
                                  __bf16* __restrict__ ob) {
  __shared__ __attribute__((aligned(16))) __bf16 kbuf[2][32 * 64];
  __shared__ __attribute__((aligned(16))) __bf16 vbuf[2][64 * 32];
  __shared__ __attribute__((aligned(16))) __bf16 pbuf[4][16 * 32];

  const int tid  = threadIdx.x;
  const int lane = tid & 31;
  const int wv   = tid >> 5;
  const int h    = blockIdx.x / (Nseq / 64);
  const int qblk = blockIdx.x % (Nseq / 64);
  const int n0   = qblk * 64 + wv * 16;

  const __bf16* qh = qb + (size_t)h * Nseq * Dh;
  const __bf16* kh = kb + (size_t)h * Nseq * Dh;
  const __bf16* vh = vT + (size_t)h * Dh * Nseq;
  __bf16* pw = &pbuf[wv][0];

  const v16bf q0 = load_frag(qh, Dh, n0, 0,  lane);
  const v16bf q1 = load_frag(qh, Dh, n0, 32, lane);

  // 16B chunk ids this thread stages (256 chunks per 4KB tile, 128 threads)
  const int c0 = tid, c1 = tid + 128;
  auto stage_tiles = [&](int jsrc, int buf) {
    unsigned kdst = (unsigned)(uintptr_t)(&kbuf[buf][0]);
    unsigned vdst = (unsigned)(uintptr_t)(&vbuf[buf][0]);
    // K tile: rows jsrc..jsrc+31 of [N][64] are one contiguous 4KB block
    async_ld_b128(kdst + c0 * 16, kh + (size_t)jsrc * Dh + c0 * 8);
    async_ld_b128(kdst + c1 * 16, kh + (size_t)jsrc * Dh + c1 * 8);
    // V^T tile: 64 rows (d) x 32 keys; source row stride Nseq
    async_ld_b128(vdst + c0 * 16,
                  vh + (size_t)(c0 >> 2) * Nseq + jsrc + (c0 & 3) * 8);
    async_ld_b128(vdst + c1 * 16,
                  vh + (size_t)(c1 >> 2) * Nseq + jsrc + (c1 & 3) * 8);
  };

  V8F o[4];
  float mrow[8], lrow[8];
#pragma unroll
  for (int j = 0; j < 8; ++j) { mrow[j] = -1.0e30f; lrow[j] = 0.f; }
#pragma unroll
  for (int t = 0; t < 4; ++t)
#pragma unroll
    for (int j = 0; j < 8; ++j) o[t].f[j] = 0.f;

  stage_tiles(0, 0);                         // preload first tile pair

  for (int j0 = 0; j0 < Nseq; j0 += 32) {
    const int cur = (j0 >> 5) & 1;
    wait_async0();                           // own staging loads landed
    __syncthreads();                         // everyone's landed; prev reads done
    if (j0 + 32 < Nseq) stage_tiles(j0 + 32, cur ^ 1);

    const __bf16* kt = &kbuf[cur][0];        // [32][64]
    const __bf16* vt = &vbuf[cur][0];        // [64][32]

    // S = Q * K^T for keys [j0, j0+32)
    v8f z = {0.f, 0.f, 0.f, 0.f, 0.f, 0.f, 0.f, 0.f};
    V8F s0, s1;
    {
      v16bf b00 = load_frag(kt, Dh, 0,  0,  lane);
      v16bf b01 = load_frag(kt, Dh, 0,  32, lane);
      v16bf b10 = load_frag(kt, Dh, 16, 0,  lane);
      v16bf b11 = load_frag(kt, Dh, 16, 32, lane);
      s0.v = wmma_bf16(q0, b00, z);
      s0.v = wmma_bf16(q1, b01, s0.v);
      s1.v = wmma_bf16(q0, b10, z);
      s1.v = wmma_bf16(q1, b11, s1.v);
    }

    // online softmax over the 32 new columns
    float sc[8];
#pragma unroll
    for (int j = 0; j < 8; ++j) {
      float v0 = s0.f[j] * SCALE;
      float v1 = s1.f[j] * SCALE;
      float tmax = redmax16(fmaxf(v0, v1));
      float nm  = fmaxf(mrow[j], tmax);
      float scl = __expf(mrow[j] - nm);
      float p0  = __expf(v0 - nm);
      float p1  = __expf(v1 - nm);
      lrow[j] = lrow[j] * scl + redsum16(p0 + p1);
      mrow[j] = nm;
      sc[j]   = scl;
      int r = j + ((lane >> 4) << 3);        // row in C-layout held by this lane
      pw[r * 32 +      (lane & 15)] = f32_to_bf16(p0);
      pw[r * 32 + 16 + (lane & 15)] = f32_to_bf16(p1);
    }
#pragma unroll
    for (int t = 0; t < 4; ++t)
#pragma unroll
      for (int j = 0; j < 8; ++j) o[t].f[j] *= sc[j];

    // O += P * V  (P re-read from LDS in A-operand layout; V^T rows as B)
    v16bf pa = load_frag(pw, 32, 0, 0, lane);
#pragma unroll
    for (int t = 0; t < 4; ++t) {
      v16bf bv = load_frag(vt, 32, t * 16, 0, lane);
      o[t].v = wmma_bf16(pa, bv, o[t].v);
    }
  }

  // normalize and store O in [N][C] bf16
#pragma unroll
  for (int t = 0; t < 4; ++t) {
#pragma unroll
    for (int j = 0; j < 8; ++j) {
      int n = n0 + j + ((lane >> 4) << 3);
      int d = t * 16 + (lane & 15);
      ob[(size_t)n * C + h * Dh + d] = f32_to_bf16(o[t].f[j] / lrow[j]);
    }
  }
}

// ---------------------------------------------------------------------------
// Output projection: out[n][o] = sum_c O[n][c] * Wproj[o][c] + b[o]  (fp32 out)
// ---------------------------------------------------------------------------
__global__ void proj_gemm_kernel(const __bf16* __restrict__ ob,
                                 const __bf16* __restrict__ wb,
                                 const float* __restrict__ bias,
                                 float* __restrict__ out) {
  const int lane = threadIdx.x & 31;
  const int wid  = blockIdx.x * (blockDim.x >> 5) + (threadIdx.x >> 5);
  const int mt = wid % (Nseq / 16);
  const int ft = wid / (Nseq / 16);          // 12 strips of 64
  const int m0 = mt * 16;
  const int f0 = ft * 64;

  V8F acc[4];
#pragma unroll
  for (int t = 0; t < 4; ++t)
#pragma unroll
    for (int j = 0; j < 8; ++j) acc[t].f[j] = 0.f;

  for (int kk = 0; kk < C; kk += 32) {
    v16bf a = load_frag(ob, C, m0, kk, lane);
#pragma unroll
    for (int t = 0; t < 4; ++t) {
      v16bf b = load_frag(wb, C, f0 + t * 16, kk, lane);
      acc[t].v = wmma_bf16(a, b, acc[t].v);
    }
  }

#pragma unroll
  for (int t = 0; t < 4; ++t) {
#pragma unroll
    for (int j = 0; j < 8; ++j) {
      int n  = m0 + j + ((lane >> 4) << 3);
      int oc = f0 + t * 16 + (lane & 15);
      out[(size_t)n * C + oc] = acc[t].f[j] + bias[oc];
    }
  }
}

// ---------------------------------------------------------------------------
extern "C" void kernel_launch(void* const* d_in, const int* in_sizes, int n_in,
                              void* d_out, int out_size, void* d_ws, size_t ws_size,
                              hipStream_t stream) {
  (void)in_sizes; (void)n_in; (void)out_size; (void)ws_size;
  const float* x     = (const float*)d_in[0];   // [1,4096,768]
  /* d_in[1] = xpos (unused: rope is None) */
  const float* Wqkv  = (const float*)d_in[2];   // [2304,768]
  const float* Wproj = (const float*)d_in[3];   // [768,768]
  const float* bproj = (const float*)d_in[4];   // [768]
  float* out = (float*)d_out;                   // [1,4096,768] fp32

  // carve 256B-aligned bf16 buffers out of d_ws (~36.2 MB total)
  unsigned char* ws = (unsigned char*)d_ws;
  size_t off = 0;
  auto take = [&](size_t elems) -> __bf16* {
    __bf16* p = (__bf16*)(ws + off);
    off += ((elems * 2 + 255) & ~(size_t)255);
    return p;
  };
  __bf16* xb  = take((size_t)Nseq * C);     // x in bf16
  __bf16* wqb = take((size_t)F3 * C);       // W_qkv bf16
  __bf16* wpb = take((size_t)C * C);        // W_proj bf16
  __bf16* qb  = take((size_t)H * Nseq * Dh);
  __bf16* kb  = take((size_t)H * Nseq * Dh);
  __bf16* vtb = take((size_t)H * Dh * Nseq);
  __bf16* ob  = take((size_t)Nseq * C);     // attention output bf16

  cvt_bf16_kernel<<<1024, 256, 0, stream>>>(x,     xb,  Nseq * C);
  cvt_bf16_kernel<<<1024, 256, 0, stream>>>(Wqkv,  wqb, F3 * C);
  cvt_bf16_kernel<<<512,  256, 0, stream>>>(Wproj, wpb, C * C);

  // QKV GEMM: 256 M-tiles x 36 strips = 9216 waves, 4 waves/block
  qkv_gemm_kernel<<<(Nseq / 16) * (F3 / 64) / 4, 128, 0, stream>>>(
      xb, wqb, qb, kb, vtb);

  // Flash attention: 12 heads x 64 query-blocks, 4 waves/block
  flash_attn_kernel<<<H * (Nseq / 64), 128, 0, stream>>>(qb, kb, vtb, ob);

  // Output projection: 256 x 12 = 3072 waves, 4 waves/block
  proj_gemm_kernel<<<(Nseq / 16) * (C / 64) / 4, 128, 0, stream>>>(
      ob, wpb, bproj, out);
}
